// D_point_plus_33706903339264
// MI455X (gfx1250) — compile-verified, matrix operand order checked by reading, not run on hardware
//
#include <hip/hip_runtime.h>
#include <hip/hip_bf16.h>

typedef __attribute__((ext_vector_type(16))) _Float16 v16h;
typedef __attribute__((ext_vector_type(8)))  _Float16 v8h;
typedef __attribute__((ext_vector_type(8)))  float    v8f;

#define BB   16
#define NPTS 4096
#define KS   64      // nsample
#define S1   512
#define S2   128

// ---------------------------------------------------------------------------
// Farthest point sampling: one block per batch, iterative argmax in LDS.
// ---------------------------------------------------------------------------
__global__ __launch_bounds__(256)
void fps_kernel(const float* __restrict__ pts, int bstride, int cstride, int pstride,
                int n, int npoint, float* __restrict__ dist,
                int* __restrict__ outIdx, float* __restrict__ outXyz) {
  const int b = blockIdx.x;
  const float* base = pts + (size_t)b * bstride;
  float* db = dist + (size_t)b * n;
  __shared__ float sVal[256];
  __shared__ int   sIdx[256];
  __shared__ int   sFar;
  for (int i = threadIdx.x; i < n; i += blockDim.x) db[i] = 1e10f;
  if (threadIdx.x == 0) sFar = 0;
  __syncthreads();
  for (int s = 0; s < npoint; ++s) {
    const int far = sFar;
    const float cx = base[(size_t)far * pstride];
    const float cy = base[(size_t)far * pstride + cstride];
    const float cz = base[(size_t)far * pstride + 2 * (size_t)cstride];
    if (threadIdx.x == 0) {
      outIdx[(size_t)b * npoint + s] = far;
      float* ox = outXyz + ((size_t)b * npoint + s) * 3;
      ox[0] = cx; ox[1] = cy; ox[2] = cz;
    }
    float best = -1.0f; int bi = 0;
    for (int i = threadIdx.x; i < n; i += blockDim.x) {
      const float dx = base[(size_t)i * pstride] - cx;
      const float dy = base[(size_t)i * pstride + cstride] - cy;
      const float dz = base[(size_t)i * pstride + 2 * (size_t)cstride] - cz;
      const float d  = dx * dx + dy * dy + dz * dz;
      const float nd = fminf(db[i], d);
      db[i] = nd;
      if (nd > best) { best = nd; bi = i; }
    }
    sVal[threadIdx.x] = best; sIdx[threadIdx.x] = bi;
    __syncthreads();
    for (int off = 128; off > 0; off >>= 1) {
      if ((int)threadIdx.x < off) {
        const float ov = sVal[threadIdx.x + off];
        const int   oi = sIdx[threadIdx.x + off];
        if (ov > sVal[threadIdx.x] || (ov == sVal[threadIdx.x] && oi < sIdx[threadIdx.x])) {
          sVal[threadIdx.x] = ov; sIdx[threadIdx.x] = oi;
        }
      }
      __syncthreads();
    }
    if (threadIdx.x == 0) sFar = sIdx[0];
    __syncthreads();
  }
}

// ---------------------------------------------------------------------------
// Ball query: first KS in-order indices with sqdist <= r2, pad with first hit.
// ---------------------------------------------------------------------------
__global__ void ball_query_kernel(const float* __restrict__ pts, int bstride, int cstride,
                                  int pstride, const float* __restrict__ centers,
                                  int S, int n, float r2, int* __restrict__ outIdx, int total) {
  const int t = blockIdx.x * blockDim.x + threadIdx.x;
  if (t >= total) return;
  const int b = t / S;
  const float* base = pts + (size_t)b * bstride;
  const float* c = centers + (size_t)t * 3;
  int* o = outIdx + (size_t)t * KS;
  int cnt = 0, first = 0;
  for (int i = 0; i < n && cnt < KS; ++i) {
    const float dx = base[(size_t)i * pstride] - c[0];
    const float dy = base[(size_t)i * pstride + cstride] - c[1];
    const float dz = base[(size_t)i * pstride + 2 * (size_t)cstride] - c[2];
    if (dx * dx + dy * dy + dz * dz <= r2) {
      if (cnt == 0) first = i;
      o[cnt++] = i;
    }
  }
  for (; cnt < KS; ++cnt) o[cnt] = first;
}

// ---------------------------------------------------------------------------
// Grouping kernels: build padded f16 GEMM inputs (zero pad channels).
// ---------------------------------------------------------------------------
__global__ void group1_kernel(const float* __restrict__ xyz /*(B,6,N)*/,
                              const float* __restrict__ centers /*(B,S1,3)*/,
                              const int* __restrict__ idx /*(B,S1,KS)*/,
                              _Float16* __restrict__ Xout /*(B*S1*KS, 32)*/) {
  const size_t total = (size_t)BB * S1 * KS * 32;
  const size_t t = (size_t)blockIdx.x * blockDim.x + threadIdx.x;
  if (t >= total) return;
  const int    c  = (int)(t & 31);
  const size_t r  = t >> 5;
  const size_t bs = r / KS;
  const int    b  = (int)(bs / S1);
  const int    j  = idx[r];
  const float* xb = xyz + (size_t)b * 6 * NPTS;
  float v = 0.0f;
  if (c < 3)      v = xb[(size_t)c * NPTS + j] - centers[bs * 3 + c];
  else if (c < 6) v = xb[(size_t)c * NPTS + j];
  Xout[t] = (_Float16)v;
}

__global__ void group2_kernel(const float* __restrict__ nxyz1 /*(B,S1,3)*/,
                              const float* __restrict__ nxyz2 /*(B,S2,3)*/,
                              const float* __restrict__ l1p /*(B,S1,128)*/,
                              const int* __restrict__ idx /*(B,S2,KS)*/,
                              _Float16* __restrict__ Xout /*(B*S2*KS, 160)*/) {
  const size_t total = (size_t)BB * S2 * KS * 160;
  const size_t t = (size_t)blockIdx.x * blockDim.x + threadIdx.x;
  if (t >= total) return;
  const int    c  = (int)(t % 160);
  const size_t r  = t / 160;
  const size_t bs = r / KS;
  const int    b  = (int)(bs / S2);
  const int    s  = (int)(bs % S2);
  const int    j  = idx[r];
  float v = 0.0f;
  if (c < 3)        v = nxyz1[((size_t)b * S1 + j) * 3 + c] - nxyz2[((size_t)b * S2 + s) * 3 + c];
  else if (c < 131) v = l1p[((size_t)b * S1 + j) * 128 + (c - 3)];
  Xout[t] = (_Float16)v;
}

__global__ void group3_kernel(const float* __restrict__ nxyz2 /*(B,S2,3)*/,
                              const float* __restrict__ l2p /*(B,S2,256)*/,
                              _Float16* __restrict__ Xout /*(B*S2, 288)*/) {
  const size_t total = (size_t)BB * S2 * 288;
  const size_t t = (size_t)blockIdx.x * blockDim.x + threadIdx.x;
  if (t >= total) return;
  const int    c = (int)(t % 288);
  const size_t r = t / 288;
  float v = 0.0f;
  if (c < 3)        v = nxyz2[r * 3 + c];
  else if (c < 259) v = l2p[r * 256 + (c - 3)];
  Xout[t] = (_Float16)v;
}

// ---------------------------------------------------------------------------
// Weight conversion: f32 (Cout,Cin) -> f16 (Cout,CinPad), zero padded.
// ---------------------------------------------------------------------------
__global__ void wcvt_kernel(const float* __restrict__ W, _Float16* __restrict__ Wh,
                            int Cin, int CinPad, int total) {
  const int t = blockIdx.x * blockDim.x + threadIdx.x;
  if (t >= total) return;
  const int ci = t % CinPad;
  const int co = t / CinPad;
  Wh[t] = (ci < Cin) ? (_Float16)W[(size_t)co * Cin + ci] : (_Float16)0.0f;
}

// ---------------------------------------------------------------------------
// WMMA GEMM: Y[M,N] f32 = X[M,K] f16 * W[N,K]^T f16 + bias[N]
// K % 32 == 0, N % 64 == 0, M % 128 == 0.
// 8 waves / block stacked in M (128x64 block tile). Each wave computes a
// 16x64 strip: one A fragment reused across 4 B fragments (4 accumulators).
// Manual double-buffering: the next K tile's 10 fragment loads are issued
// before the current tile's 4 WMMAs and stay live across them (copy-down at
// loop bottom), so the WMMAs only depend on loads issued one iteration ago
// and the scheduler can use partial s_wait_loadcnt instead of full drains.
// Last iteration clamps kn to 0 (branch-free; redundant load, no OOB).
// Fragment addressing follows CDNA5 ISA 7.12.2.
// ---------------------------------------------------------------------------
__global__ __launch_bounds__(256)
void gemm_wmma_f16(const _Float16* __restrict__ X, const _Float16* __restrict__ Wt,
                   const float* __restrict__ bias, float* __restrict__ Y,
                   int M, int K, int N) {
  const int lane  = threadIdx.x & 31;
  const int wave  = threadIdx.x >> 5;
  const int h     = lane >> 4;        // half-wave: selects K sub-block
  const int l16   = lane & 15;
  const int mBase = blockIdx.x * 128 + wave * 16;
  const int nBase = blockIdx.y * 64;

  const _Float16* xp  = X  + (size_t)(mBase + l16) * K + h * 8;
  const _Float16* wp0 = Wt + (size_t)(nBase + l16) * K + h * 8;
  const size_t    wstep = (size_t)16 * K;   // next 16-column group of W

  v8f acc[4] = {{}, {}, {}, {}};

  // Prologue: load K-tile 0 into the "current" fragment set.
  v8h a0c = *(const v8h*)(xp);
  v8h a1c = *(const v8h*)(xp + 16);
  v8h b0c[4], b1c[4];
  #pragma unroll
  for (int nb = 0; nb < 4; ++nb) {
    const _Float16* wp = wp0 + (size_t)nb * wstep;
    b0c[nb] = *(const v8h*)(wp);
    b1c[nb] = *(const v8h*)(wp + 16);
  }

  for (int k = 0; k < K; k += 32) {
    const int kn = (k + 32 < K) ? (k + 32) : 0;   // clamp: uniform, branch-free
    __builtin_prefetch(xp + kn + 32, 0, 1);       // -> global_prefetch_b8
    // ---- issue next tile's loads (kept live across the WMMA batch) ----
    const v8h a0n = *(const v8h*)(xp + kn);
    const v8h a1n = *(const v8h*)(xp + kn + 16);
    v8h b0n[4], b1n[4];
    #pragma unroll
    for (int nb = 0; nb < 4; ++nb) {
      const _Float16* wp = wp0 + (size_t)nb * wstep + kn;
      b0n[nb] = *(const v8h*)(wp);
      b1n[nb] = *(const v8h*)(wp + 16);
    }
    // ---- compute on the current tile: 4 WMMAs sharing the A fragment ----
    const v16h av = __builtin_shufflevector(a0c, a1c,
        0,1,2,3,4,5,6,7,8,9,10,11,12,13,14,15);
    #pragma unroll
    for (int nb = 0; nb < 4; ++nb) {
      const v16h bv = __builtin_shufflevector(b0c[nb], b1c[nb],
          0,1,2,3,4,5,6,7,8,9,10,11,12,13,14,15);
      acc[nb] = __builtin_amdgcn_wmma_f32_16x16x32_f16(false, av, false, bv,
                                                       (short)0, acc[nb], false, false);
    }
    // ---- copy-down ----
    a0c = a0n; a1c = a1n;
    #pragma unroll
    for (int nb = 0; nb < 4; ++nb) { b0c[nb] = b0n[nb]; b1c[nb] = b1n[nb]; }
  }

  #pragma unroll
  for (int nb = 0; nb < 4; ++nb) {
    const int col = nBase + nb * 16 + l16;
    const float bsum = bias[col];
    #pragma unroll
    for (int v = 0; v < 8; ++v) {
      const int row = mBase + v + 8 * h;   // C/D layout: lane half selects M+8
      Y[(size_t)row * N + col] = acc[nb][v] + bsum;
    }
  }
}

// ---------------------------------------------------------------------------
// BatchNorm over all M rows per channel: two-stage (partials via atomics).
// ---------------------------------------------------------------------------
__global__ void stats_init_kernel(float* __restrict__ st, int n) {
  const int t = blockIdx.x * blockDim.x + threadIdx.x;
  if (t < n) st[t] = 0.0f;
}

__global__ __launch_bounds__(256)
void stats_kernel(const float* __restrict__ Y, int M, int C, float* __restrict__ st) {
  const int rpb = (M + gridDim.x - 1) / gridDim.x;
  const int r0 = blockIdx.x * rpb;
  const int r1 = (r0 + rpb < M) ? (r0 + rpb) : M;
  float s[4] = {0, 0, 0, 0}, q[4] = {0, 0, 0, 0};
  for (int r = r0; r < r1; ++r) {
    const float* row = Y + (size_t)r * C;
    int j = 0;
    for (int c = threadIdx.x; c < C; c += blockDim.x, ++j) {
      const float v = row[c];
      s[j] += v; q[j] += v * v;
    }
  }
  int j = 0;
  for (int c = threadIdx.x; c < C; c += blockDim.x, ++j) {
    atomicAdd(&st[c], s[j]);
    atomicAdd(&st[C + c], q[j]);
  }
}

__global__ void bn_relu_kernel(const float* __restrict__ Y, const float* __restrict__ st,
                               const float* __restrict__ gam, const float* __restrict__ bet,
                               _Float16* __restrict__ Xout, int M, int C) {
  const size_t total = (size_t)M * C;
  const size_t t = (size_t)blockIdx.x * blockDim.x + threadIdx.x;
  if (t >= total) return;
  const int c = (int)(t % C);
  const float invM = 1.0f / (float)M;
  const float mu   = st[c] * invM;
  const float var  = st[C + c] * invM - mu * mu;
  const float v = (Y[t] - mu) * rsqrtf(var + 1e-5f) * gam[c] + bet[c];
  Xout[t] = (_Float16)fmaxf(v, 0.0f);
}

// ---------------------------------------------------------------------------
// Max pool over the nsample axis: (BS, K, C) f16 -> (BS, C) f32.
// ---------------------------------------------------------------------------
__global__ void maxpool_kernel(const _Float16* __restrict__ Xin, float* __restrict__ out,
                               int BS, int K, int C) {
  const int t = blockIdx.x * blockDim.x + threadIdx.x;
  if (t >= BS * C) return;
  const int c  = t % C;
  const int bs = t / C;
  const _Float16* p = Xin + (size_t)bs * K * C + c;
  float m = -1e30f;
  for (int k = 0; k < K; ++k) m = fmaxf(m, (float)p[(size_t)k * C]);
  out[t] = m;
}

// ---------------------------------------------------------------------------
// FC head (tiny, f32 scalar).
// ---------------------------------------------------------------------------
__global__ void fc_kernel(const float* __restrict__ x, const float* __restrict__ W,
                          const float* __restrict__ b, float* __restrict__ y,
                          int Cin, int Cout) {
  const int t = blockIdx.x * blockDim.x + threadIdx.x;
  if (t >= BB * Cout) return;
  const int o = t % Cout;
  const int r = t / Cout;
  const float* xr = x + (size_t)r * Cin;
  const float* wr = W + (size_t)o * Cin;
  float acc = b[o];
  for (int i = 0; i < Cin; ++i) acc += xr[i] * wr[i];
  y[t] = acc;
}

__global__ void bn1d_relu_kernel(float* __restrict__ x, const float* __restrict__ g,
                                 const float* __restrict__ b, int C) {
  const int c = blockIdx.x * blockDim.x + threadIdx.x;
  if (c >= C) return;
  float s = 0.0f, q = 0.0f;
  for (int r = 0; r < BB; ++r) { const float v = x[(size_t)r * C + c]; s += v; q += v * v; }
  const float mu  = s / BB;
  const float var = q / BB - mu * mu;
  const float inv = rsqrtf(var + 1e-5f) * g[c];
  for (int r = 0; r < BB; ++r) {
    const float v = (x[(size_t)r * C + c] - mu) * inv + b[c];
    x[(size_t)r * C + c] = fmaxf(v, 0.0f);
  }
}

__global__ void writeout_kernel(const float* __restrict__ fcc, float* __restrict__ out) {
  const int t = threadIdx.x;
  if (t < BB) { out[t] = fcc[t]; out[BB + t] = fcc[t]; out[2 * BB + t] = fcc[t]; }
}

// ---------------------------------------------------------------------------
// Host orchestration.
// ---------------------------------------------------------------------------
extern "C" void kernel_launch(void* const* d_in, const int* in_sizes, int n_in,
                              void* d_out, int out_size, void* d_ws, size_t ws_size,
                              hipStream_t stream) {
  // ---- input index mapping (insertion-order vs sorted-key flattening) ----
  int SA[3][3][4], FC[3][2], BN[2][2], XYZ;
  const bool insertion = (n_in > 0 && in_sizes[0] == BB * 6 * NPTS);
  if (insertion) {
    XYZ = 0;
    for (int s = 0; s < 3; ++s) for (int l = 0; l < 3; ++l) for (int j = 0; j < 4; ++j)
      SA[s][l][j] = 1 + s * 12 + l * 4 + j;
    for (int f = 0; f < 3; ++f) for (int j = 0; j < 2; ++j) FC[f][j] = 37 + f * 2 + j;
    for (int b = 0; b < 2; ++b) for (int j = 0; j < 2; ++j) BN[b][j] = 43 + b * 2 + j;
  } else {
    for (int b = 0; b < 2; ++b) for (int j = 0; j < 2; ++j) BN[b][j] = b * 2 + j;
    for (int f = 0; f < 3; ++f) for (int j = 0; j < 2; ++j) FC[f][j] = 4 + f * 2 + j;
    for (int s = 0; s < 3; ++s) for (int l = 0; l < 3; ++l) for (int j = 0; j < 4; ++j)
      SA[s][l][j] = 10 + s * 12 + l * 4 + j;
    XYZ = 46;
  }
  const float* xyz = (const float*)d_in[XYZ];

  // ---- workspace carve-up ----
  char* ws = (char*)d_ws;
  size_t off = 0;
  auto alloc = [&](size_t bytes) -> char* {
    char* p = ws + off;
    off = (off + bytes + 255) & ~(size_t)255;
    return p;
  };
  float*    dist  = (float*)alloc((size_t)BB * NPTS * 4);
  int*      fidx1 = (int*)  alloc((size_t)BB * S1 * 4);
  float*    nxyz1 = (float*)alloc((size_t)BB * S1 * 3 * 4);
  int*      bidx1 = (int*)  alloc((size_t)BB * S1 * KS * 4);
  int*      fidx2 = (int*)  alloc((size_t)BB * S2 * 4);
  float*    nxyz2 = (float*)alloc((size_t)BB * S2 * 3 * 4);
  int*      bidx2 = (int*)  alloc((size_t)BB * S2 * KS * 4);
  float*    l1p   = (float*)alloc((size_t)BB * S1 * 128 * 4);
  float*    l2p   = (float*)alloc((size_t)BB * S2 * 256 * 4);
  float*    l3p   = (float*)alloc((size_t)BB * 1024 * 4);
  float*    stats = (float*)alloc(2 * 1024 * 4);
  _Float16* wh    = (_Float16*)alloc((size_t)1048576 * 2);
  float*    fca   = (float*)alloc((size_t)BB * 512 * 4);
  float*    fcb   = (float*)alloc((size_t)BB * 256 * 4);
  float*    fcc   = (float*)alloc((size_t)BB * 4);
  _Float16* Xbuf  = (_Float16*)alloc((size_t)524288 * 128 * 2);
  float*    Ybuf  = (float*)alloc((size_t)524288 * 128 * 4);
  (void)ws_size;

  auto cdiv = [](size_t a, size_t b) -> unsigned { return (unsigned)((a + b - 1) / b); };

  // Run one MLP layer: wcvt -> WMMA GEMM -> BN stats -> BN+ReLU (writes f16).
  auto run_layer = [&](int sa, int l, int M, int Cin, int CinPad, int Cout) {
    const float* W   = (const float*)d_in[SA[sa][l][0]];
    const float* bia = (const float*)d_in[SA[sa][l][1]];
    const float* gam = (const float*)d_in[SA[sa][l][2]];
    const float* bet = (const float*)d_in[SA[sa][l][3]];
    const int wtot = Cout * CinPad;
    wcvt_kernel<<<cdiv(wtot, 256), 256, 0, stream>>>(W, wh, Cin, CinPad, wtot);
    dim3 gg(M / 128, Cout / 64);
    gemm_wmma_f16<<<gg, 256, 0, stream>>>(Xbuf, wh, bia, Ybuf, M, CinPad, Cout);
    stats_init_kernel<<<cdiv(2 * Cout, 256), 256, 0, stream>>>(stats, 2 * Cout);
    const unsigned sgrid = (M >= 262144) ? 1024u : cdiv(M, 256);
    stats_kernel<<<sgrid, 256, 0, stream>>>(Ybuf, M, Cout, stats);
    bn_relu_kernel<<<cdiv((size_t)M * Cout, 256), 256, 0, stream>>>(
        Ybuf, stats, gam, bet, Xbuf, M, Cout);
  };

  // ================= SA1 =================
  const int M1 = BB * S1 * KS;  // 524288
  fps_kernel<<<BB, 256, 0, stream>>>(xyz, 6 * NPTS, NPTS, 1, NPTS, S1, dist, fidx1, nxyz1);
  ball_query_kernel<<<cdiv(BB * S1, 256), 256, 0, stream>>>(
      xyz, 6 * NPTS, NPTS, 1, nxyz1, S1, NPTS, 0.015f * 0.015f, bidx1, BB * S1);
  group1_kernel<<<cdiv((size_t)M1 * 32, 256), 256, 0, stream>>>(xyz, nxyz1, bidx1, Xbuf);
  run_layer(0, 0, M1, 6, 32, 64);
  run_layer(0, 1, M1, 64, 64, 64);
  run_layer(0, 2, M1, 64, 64, 128);
  maxpool_kernel<<<cdiv(BB * S1 * 128, 256), 256, 0, stream>>>(Xbuf, l1p, BB * S1, KS, 128);

  // ================= SA2 =================
  const int M2 = BB * S2 * KS;  // 131072
  fps_kernel<<<BB, 256, 0, stream>>>(nxyz1, S1 * 3, 1, 3, S1, S2, dist, fidx2, nxyz2);
  ball_query_kernel<<<cdiv(BB * S2, 256), 256, 0, stream>>>(
      nxyz1, S1 * 3, 1, 3, nxyz2, S2, S1, 0.04f * 0.04f, bidx2, BB * S2);
  group2_kernel<<<cdiv((size_t)M2 * 160, 256), 256, 0, stream>>>(nxyz1, nxyz2, l1p, bidx2, Xbuf);
  run_layer(1, 0, M2, 131, 160, 128);
  run_layer(1, 1, M2, 128, 128, 128);
  run_layer(1, 2, M2, 128, 128, 256);
  maxpool_kernel<<<cdiv(BB * S2 * 256, 256), 256, 0, stream>>>(Xbuf, l2p, BB * S2, KS, 256);

  // ================= SA3 (group_all) =================
  const int M3 = BB * S2;  // 2048
  group3_kernel<<<cdiv((size_t)M3 * 288, 256), 256, 0, stream>>>(nxyz2, l2p, Xbuf);
  run_layer(2, 0, M3, 259, 288, 256);
  run_layer(2, 1, M3, 256, 256, 512);
  run_layer(2, 2, M3, 512, 512, 1024);
  maxpool_kernel<<<cdiv(BB * 1024, 256), 256, 0, stream>>>(Xbuf, l3p, BB, S2, 1024);

  // ================= FC head =================
  fc_kernel<<<cdiv(BB * 512, 256), 256, 0, stream>>>(
      l3p, (const float*)d_in[FC[0][0]], (const float*)d_in[FC[0][1]], fca, 1024, 512);
  bn1d_relu_kernel<<<cdiv(512, 256), 256, 0, stream>>>(
      fca, (const float*)d_in[BN[0][0]], (const float*)d_in[BN[0][1]], 512);
  fc_kernel<<<cdiv(BB * 256, 256), 256, 0, stream>>>(
      fca, (const float*)d_in[FC[1][0]], (const float*)d_in[FC[1][1]], fcb, 512, 256);
  bn1d_relu_kernel<<<cdiv(256, 256), 256, 0, stream>>>(
      fcb, (const float*)d_in[BN[1][0]], (const float*)d_in[BN[1][1]], 256);
  fc_kernel<<<cdiv(BB * 1, 256), 256, 0, stream>>>(
      fcb, (const float*)d_in[FC[2][0]], (const float*)d_in[FC[2][1]], fcc, 256, 1);

  writeout_kernel<<<1, 64, 0, stream>>>(fcc, (float*)d_out);
  (void)out_size;
}
